// LinearAttnNode_71562745086262
// MI455X (gfx1250) — matrix operand order repaired and checked by reading, not compiled, not run on hardware
//
#include <hip/hip_runtime.h>
#include <hip/hip_bf16.h>
#include <math.h>

// ---------------------------------------------------------------------------
// Causal linear attention (random Fourier features) for MI455X / gfx1250.
// All GEMM-shaped work runs through v_wmma_f32_16x16x32_f16 (f16 in, f32 acc).
// ---------------------------------------------------------------------------

typedef _Float16 h16;
typedef __attribute__((ext_vector_type(16))) _Float16 v16h;
typedef __attribute__((ext_vector_type(8)))  float    v8f;

#define WMMA_F16(a, b, c) \
  __builtin_amdgcn_wmma_f32_16x16x32_f16(false, (a), false, (b), (short)0, (c), false, false)

constexpr int Bb = 2, Tt = 1024, Dd = 512, Hh = 8, HD = 64, Ff = 64, F2 = 128;
constexpr int Mrows = Bb * Tt;  // 2048

// --- Fragment gathers (wave32), contiguous k order shared by A and B ----------
// A (16x32 f16): lane l holds row m=l&15; element j: k = (l>>4)*16 + j
__device__ inline v16h a_frag_lds(const h16* base, int ld) {
  int l = threadIdx.x & 31;
  int m = l & 15, hs = l >> 4;
  v16h a;
#pragma unroll
  for (int j = 0; j < 16; ++j) {
    int k = (hs << 4) + j;
    a[j] = base[m * ld + k];
  }
  return a;
}
// B (32x16 f16): lane l holds column n=l&15; lanes 0-15 K=0..15, lanes 16-31 K=16..31
__device__ inline v16h b_frag_lds(const h16* base, int ld) {
  int l = threadIdx.x & 31;
  int n = l & 15, kb = (l >> 4) << 4;
  v16h b;
#pragma unroll
  for (int j = 0; j < 16; ++j) b[j] = base[(kb + j) * ld + n];
  return b;
}
// C/D (16x16 f32, 8 VGPRs): element r -> (m = (l>>4)*8 + r, n = l&15)

// ---------------------------------------------------------------------------
// Kernel 1: RMSNorm. One block per (b,t) row of 512.
// ---------------------------------------------------------------------------
__global__ void k_rmsnorm(const float* __restrict__ x, const float* __restrict__ w,
                          float* __restrict__ xn) {
  __shared__ float red[256];
  int row = blockIdx.x;
  const float* xr = x + (size_t)row * Dd;
  float v0 = xr[threadIdx.x], v1 = xr[threadIdx.x + 256];
  red[threadIdx.x] = v0 * v0 + v1 * v1;
  __syncthreads();
  for (int off = 128; off > 0; off >>= 1) {
    if (threadIdx.x < off) red[threadIdx.x] += red[threadIdx.x + off];
    __syncthreads();
  }
  float scale = rsqrtf(red[0] / (float)Dd + 1e-6f);
  xn[(size_t)row * Dd + threadIdx.x]       = v0 * scale * w[threadIdx.x];
  xn[(size_t)row * Dd + threadIdx.x + 256] = v1 * scale * w[threadIdx.x + 256];
}

// ---------------------------------------------------------------------------
// Kernel 2: q/k/v projection GEMM.  C[m,n] = sum_k xn[m,k]*W[n,k] + bias[n].
// 64x64 output tile per block (8 waves x 2 WMMA tiles), K swept in 32 steps.
// Epilogue scatters into [B,H,T,HD] f16.
// ---------------------------------------------------------------------------
__global__ void k_proj_qkv(const float* __restrict__ xn, const float* __restrict__ W,
                           const float* __restrict__ bias, h16* __restrict__ out) {
  __shared__ h16 As[64][34];
  __shared__ h16 Bs[32][66];
  int tid = threadIdx.x, lane = tid & 31, wave = tid >> 5;
  int n0 = blockIdx.x * 64, m0 = blockIdx.y * 64;
  int tr = wave & 3, tc0 = (wave >> 2) * 2;
  v8f acc0 = {}, acc1 = {};
  for (int kk = 0; kk < Dd; kk += 32) {
    {  // A tile (f32 -> f16 during staging)
      int row = tid >> 2, c0 = (tid & 3) * 8;
      const float* s = xn + (size_t)(m0 + row) * Dd + kk + c0;
#pragma unroll
      for (int u = 0; u < 8; ++u) As[row][c0 + u] = (h16)s[u];
    }
    {  // B tile: Bs[k][n] = W[n0+n][kk+k]
      int n = tid >> 2, c0 = (tid & 3) * 8;
      const float* s = W + (size_t)(n0 + n) * Dd + kk + c0;
#pragma unroll
      for (int u = 0; u < 8; ++u) Bs[c0 + u][n] = (h16)s[u];
    }
    __syncthreads();
    v16h a  = a_frag_lds(&As[tr * 16][0], 34);
    v16h b0 = b_frag_lds(&Bs[0][tc0 * 16], 66);
    v16h b1 = b_frag_lds(&Bs[0][(tc0 + 1) * 16], 66);
    acc0 = WMMA_F16(a, b0, acc0);
    acc1 = WMMA_F16(a, b1, acc1);
    __syncthreads();
  }
#pragma unroll
  for (int r = 0; r < 8; ++r) {
    int mrow = m0 + tr * 16 + ((lane >> 4) << 3) + r;
    int bb = mrow >> 10, t = mrow & 1023;
#pragma unroll
    for (int p = 0; p < 2; ++p) {
      int ncol = n0 + (tc0 + p) * 16 + (lane & 15);
      float val = (p ? acc1[r] : acc0[r]) + bias[ncol];
      int h = ncol >> 6, hd = ncol & 63;
      out[((size_t)(bb * Hh + h) * Tt + t) * HD + hd] = (h16)val;
    }
  }
}

// ---------------------------------------------------------------------------
// Kernel 3: feature map.  proj = head @ omega^T, out = [sin(proj), cos(proj)].
// blockIdx.z selects q (0) or k (1).
// ---------------------------------------------------------------------------
__global__ void k_phi(const h16* __restrict__ qh, const h16* __restrict__ kh,
                      const float* __restrict__ omega,
                      h16* __restrict__ qf, h16* __restrict__ kf) {
  __shared__ h16 As[64][34];
  __shared__ h16 Bs[32][66];
  const h16* src = (blockIdx.z == 0) ? qh : kh;
  h16* dst       = (blockIdx.z == 0) ? qf : kf;
  int tid = threadIdx.x, lane = tid & 31, wave = tid >> 5;
  int bh = blockIdx.y;
  int t0 = blockIdx.x * 64;
  int tr = wave & 3, tc0 = (wave >> 2) * 2;
  v8f acc0 = {}, acc1 = {};
  const h16* abase = src + ((size_t)bh * Tt + t0) * HD;
  for (int kk = 0; kk < HD; kk += 32) {
    { int row = tid >> 2, c0 = (tid & 3) * 8;
      const h16* s = abase + (size_t)row * HD + kk + c0;
#pragma unroll
      for (int u = 0; u < 8; ++u) As[row][c0 + u] = s[u]; }
    { int n = tid >> 2, c0 = (tid & 3) * 8;   // Bs[k][n] = omega[f=n][hd=kk+k]
      const float* s = omega + (size_t)n * HD + kk + c0;
#pragma unroll
      for (int u = 0; u < 8; ++u) Bs[c0 + u][n] = (h16)s[u]; }
    __syncthreads();
    v16h a  = a_frag_lds(&As[tr * 16][0], 34);
    v16h b0 = b_frag_lds(&Bs[0][tc0 * 16], 66);
    v16h b1 = b_frag_lds(&Bs[0][(tc0 + 1) * 16], 66);
    acc0 = WMMA_F16(a, b0, acc0);
    acc1 = WMMA_F16(a, b1, acc1);
    __syncthreads();
  }
  h16* obase = dst + ((size_t)bh * Tt + t0) * F2;
#pragma unroll
  for (int r = 0; r < 8; ++r) {
    int mrow = tr * 16 + ((lane >> 4) << 3) + r;
#pragma unroll
    for (int p = 0; p < 2; ++p) {
      int f = (tc0 + p) * 16 + (lane & 15);
      float pr = p ? acc1[r] : acc0[r];
      obase[(size_t)mrow * F2 + f]      = (h16)sinf(pr);
      obase[(size_t)mrow * F2 + Ff + f] = (h16)cosf(pr);
    }
  }
}

// ---------------------------------------------------------------------------
// Kernel 4: chunked causal linear-attention scan. One block per (b,h).
// State S[128][64] f32 in LDS; chunks of 16 timesteps; every matrix product
// (Qf*Kf^T, Qf*S, maskedA*V, Kf^T*V) is a WMMA (K padded to 32 where needed).
// ---------------------------------------------------------------------------
__global__ void k_scan(const h16* __restrict__ qf, const h16* __restrict__ kf,
                       const h16* __restrict__ vv, h16* __restrict__ attn) {
  __shared__ float S[F2][HD];     // 32 KB running state
  __shared__ float zf[F2];
  __shared__ h16 Qs[16][F2];
  __shared__ h16 Ks[16][F2];
  __shared__ h16 Vs[16][HD];
  __shared__ h16 Ah[16][32];      // masked scores, cols 16..31 stay zero (K pad)
  __shared__ float den_s[16];

  int tid = threadIdx.x, lane = tid & 31, wave = tid >> 5;
  int bh = blockIdx.x;
  for (int i = tid; i < F2 * HD; i += 256) S[i / HD][i % HD] = 0.f;
  if (tid < F2) zf[tid] = 0.f;
  for (int i = tid; i < 16 * 32; i += 256) Ah[i >> 5][i & 31] = (h16)0.f;
  __syncthreads();

  const h16* qbase = qf + (size_t)bh * Tt * F2;
  const h16* kbase = kf + (size_t)bh * Tt * F2;
  const h16* vbase = vv + (size_t)bh * Tt * HD;
  h16*       obase = attn + (size_t)bh * Tt * HD;

  for (int c = 0; c < Tt / 16; ++c) {
    int t0 = c * 16;
#pragma unroll
    for (int u = 0; u < 8; ++u) {  // 2048 halves each
      int idx = tid * 8 + u, row = idx >> 7, col = idx & 127;
      Qs[row][col] = qbase[(size_t)(t0 + row) * F2 + col];
      Ks[row][col] = kbase[(size_t)(t0 + row) * F2 + col];
    }
#pragma unroll
    for (int u = 0; u < 4; ++u) {  // 1024 halves
      int idx = tid * 4 + u, row = idx >> 6, col = idx & 63;
      Vs[row][col] = vbase[(size_t)(t0 + row) * HD + col];
    }
    __syncthreads();

    // Phase A: wave0 -> A = Qf Kf^T (masked); waves 4-7 -> Kf^T V accumulators
    v8f accK[2][4];
    if (wave >= 4) {
      v8f z = {};
#pragma unroll
      for (int fi = 0; fi < 2; ++fi)
#pragma unroll
        for (int dt = 0; dt < 4; ++dt) accK[fi][dt] = z;
      int m = lane & 15, hs = lane >> 4, kb = hs << 4;
#pragma unroll
      for (int fi = 0; fi < 2; ++fi) {
        int ft = (wave - 4) * 2 + fi;
        v16h a;  // A[m=f_local][k=t] = Ks[t][ft*16+m], K padded 16->32
#pragma unroll
        for (int j = 0; j < 16; ++j) {
          int k = kb + j;
          a[j] = (k < 16) ? Ks[k][ft * 16 + m] : (h16)0.f;
        }
#pragma unroll
        for (int dt = 0; dt < 4; ++dt) {
          v16h b;  // B[k=t][n=d] = Vs[t][dt*16+n], padded
#pragma unroll
          for (int j = 0; j < 16; ++j) {
            int k = kb + j;
            b[j] = (k < 16) ? Vs[k][dt * 16 + (lane & 15)] : (h16)0.f;
          }
          accK[fi][dt] = WMMA_F16(a, b, accK[fi][dt]);
        }
      }
    } else if (wave == 0) {
      v8f accA = {};
      int n = lane & 15, kb0 = (lane >> 4) << 4;
#pragma unroll
      for (int it = 0; it < 4; ++it) {  // K = 128 over 4 WMMAs
        v16h a = a_frag_lds(&Qs[0][it * 32], F2);
        v16h b;  // B[k=f][n=s] = Ks[s][it*32 + k]  (Kf transposed)
#pragma unroll
        for (int j = 0; j < 16; ++j) b[j] = Ks[n][it * 32 + kb0 + j];
        accA = WMMA_F16(a, b, accA);
      }
#pragma unroll
      for (int r = 0; r < 8; ++r) {  // causal mask within chunk, store f16
        int m = ((lane >> 4) << 3) + r;
        Ah[m][n] = (h16)((n <= m) ? accA[r] : 0.f);
      }
    }
    __syncthreads();

    // Phase B: waves 0-3 -> num = Qf@S_prev + maskedA@V ; wave4 lanes<16 -> den
    v8f accN = {};
    if (wave < 4) {
      int dt = wave, n = lane & 15, kb0 = (lane >> 4) << 4;
#pragma unroll
      for (int it = 0; it < 4; ++it) {
        v16h a = a_frag_lds(&Qs[0][it * 32], F2);
        v16h b;  // B[k=f][n=d] = (f16)S[f][dt*16+n]  (convert on gather)
#pragma unroll
        for (int j = 0; j < 16; ++j) b[j] = (h16)S[it * 32 + kb0 + j][dt * 16 + n];
        accN = WMMA_F16(a, b, accN);
      }
      v16h a2 = a_frag_lds(&Ah[0][0], 32);
      v16h b2;
#pragma unroll
      for (int j = 0; j < 16; ++j) {
        int k = kb0 + j;
        b2[j] = (k < 16) ? Vs[k][dt * 16 + n] : (h16)0.f;
      }
      accN = WMMA_F16(a2, b2, accN);
    } else if (wave == 4 && lane < 16) {
      int t = lane;
      float d = 0.f;
      for (int f = 0; f < F2; ++f) d += (float)Qs[t][f] * zf[f];
      for (int s2 = 0; s2 <= t; ++s2) d += (float)Ah[t][s2];
      den_s[t] = fmaxf(d, 1e-6f);
    }
    __syncthreads();

    // Phase C: waves 0-3 epilogue (num/den -> global); waves 4-7 update S, z
    if (wave < 4) {
      int dt = wave, n = lane & 15;
#pragma unroll
      for (int r = 0; r < 8; ++r) {
        int m = ((lane >> 4) << 3) + r;
        obase[(size_t)(t0 + m) * HD + dt * 16 + n] = (h16)(accN[r] / den_s[m]);
      }
    } else {
      int n = lane & 15;
#pragma unroll
      for (int fi = 0; fi < 2; ++fi) {
        int ft = (wave - 4) * 2 + fi;
#pragma unroll
        for (int dt = 0; dt < 4; ++dt)
#pragma unroll
          for (int r = 0; r < 8; ++r) {
            int m = ((lane >> 4) << 3) + r;
            S[ft * 16 + m][dt * 16 + n] += accK[fi][dt][r];
          }
      }
      int f = tid - 128;  // 128 threads cover z exactly
      float s = 0.f;
#pragma unroll
      for (int t = 0; t < 16; ++t) s += (float)Ks[t][f];
      zf[f] += s;
    }
    __syncthreads();
  }
}

// ---------------------------------------------------------------------------
// Kernel 5: output projection + residual.  out = attn_flat @ Wo^T + bo + xn.
// A gathered from head-interleaved attn layout during staging.
// ---------------------------------------------------------------------------
__global__ void k_out_proj(const h16* __restrict__ attn, const float* __restrict__ Wo,
                           const float* __restrict__ bo, const float* __restrict__ xn,
                           float* __restrict__ out) {
  __shared__ h16 As[64][34];
  __shared__ h16 Bs[32][66];
  int tid = threadIdx.x, lane = tid & 31, wave = tid >> 5;
  int n0 = blockIdx.x * 64, m0 = blockIdx.y * 64;
  int tr = wave & 3, tc0 = (wave >> 2) * 2;
  v8f acc0 = {}, acc1 = {};
  for (int kk = 0; kk < Dd; kk += 32) {
    { int row = tid >> 2, c0 = (tid & 3) * 8;
      int mrow = m0 + row, bb = mrow >> 10, t = mrow & 1023;
#pragma unroll
      for (int u = 0; u < 8; ++u) {
        int kidx = kk + c0 + u, h = kidx >> 6, hd = kidx & 63;
        As[row][c0 + u] = attn[((size_t)(bb * Hh + h) * Tt + t) * HD + hd];
      } }
    { int n = tid >> 2, c0 = (tid & 3) * 8;
      const float* s = Wo + (size_t)(n0 + n) * Dd + kk + c0;
#pragma unroll
      for (int u = 0; u < 8; ++u) Bs[c0 + u][n] = (h16)s[u]; }
    __syncthreads();
    v16h a  = a_frag_lds(&As[tr * 16][0], 34);
    v16h b0 = b_frag_lds(&Bs[0][tc0 * 16], 66);
    v16h b1 = b_frag_lds(&Bs[0][(tc0 + 1) * 16], 66);
    acc0 = WMMA_F16(a, b0, acc0);
    acc1 = WMMA_F16(a, b1, acc1);
    __syncthreads();
  }
#pragma unroll
  for (int r = 0; r < 8; ++r) {
    int mrow = m0 + tr * 16 + ((lane >> 4) << 3) + r;
#pragma unroll
    for (int p = 0; p < 2; ++p) {
      int ncol = n0 + (tc0 + p) * 16 + (lane & 15);
      float val = (p ? acc1[r] : acc0[r]) + bo[ncol] + xn[(size_t)mrow * Dd + ncol];
      out[(size_t)mrow * Dd + ncol] = val;
    }
  }
}

// ---------------------------------------------------------------------------
extern "C" void kernel_launch(void* const* d_in, const int* in_sizes, int n_in,
                              void* d_out, int out_size, void* d_ws, size_t ws_size,
                              hipStream_t stream) {
  (void)in_sizes; (void)n_in; (void)out_size; (void)ws_size;
  const float* x      = (const float*)d_in[0];
  const float* norm_w = (const float*)d_in[1];
  const float* Wq = (const float*)d_in[2];
  const float* bq = (const float*)d_in[3];
  const float* Wk = (const float*)d_in[4];
  const float* bk = (const float*)d_in[5];
  const float* Wv = (const float*)d_in[6];
  const float* bv = (const float*)d_in[7];
  const float* Wo = (const float*)d_in[8];
  const float* bo = (const float*)d_in[9];
  const float* omega = (const float*)d_in[10];
  float* out = (float*)d_out;

  char* ws = (char*)d_ws;
  float* xn = (float*)ws;            ws += (size_t)Mrows * Dd * sizeof(float);
  h16* q    = (h16*)ws;              ws += (size_t)Mrows * Dd * sizeof(h16);
  h16* k    = (h16*)ws;              ws += (size_t)Mrows * Dd * sizeof(h16);
  h16* v    = (h16*)ws;              ws += (size_t)Mrows * Dd * sizeof(h16);
  h16* qf   = (h16*)ws;              ws += (size_t)Bb * Hh * Tt * F2 * sizeof(h16);
  h16* kf   = (h16*)ws;              ws += (size_t)Bb * Hh * Tt * F2 * sizeof(h16);
  h16* attn = (h16*)ws;

  k_rmsnorm<<<Mrows, 256, 0, stream>>>(x, norm_w, xn);

  dim3 gGemm(Dd / 64, Mrows / 64);  // (8, 32)
  k_proj_qkv<<<gGemm, 256, 0, stream>>>(xn, Wq, bq, q);
  k_proj_qkv<<<gGemm, 256, 0, stream>>>(xn, Wk, bk, k);
  k_proj_qkv<<<gGemm, 256, 0, stream>>>(xn, Wv, bv, v);

  dim3 gPhi(Tt / 64, Bb * Hh, 2);   // (16, 16, 2)
  k_phi<<<gPhi, 256, 0, stream>>>(q, k, omega, qf, kf);

  k_scan<<<Bb * Hh, 256, 0, stream>>>(qf, kf, v, attn);

  k_out_proj<<<gGemm, 256, 0, stream>>>(attn, Wo, bo, xn, out);
}
